// GraphSAGE_85134841741498
// MI455X (gfx1250) — matrix operand order repaired and checked by reading, not compiled, not run on hardware
//
#include <hip/hip_runtime.h>

// GraphSAGE on MI455X (gfx1250):
//  - mean aggregation: per-edge wave gather (bf16 rows) + fp32 global_atomic_add_f32
//    at explicit device scope into an L2-resident (102MB < 192MB L2) accumulator
//  - GEMMs: v_wmma_f32_16x16x32_bf16, fp32 accumulate, bias+PReLU fused
//  - activations stored bf16 between layers; all accumulation fp32

typedef __bf16 bf16;
typedef __attribute__((ext_vector_type(16))) __bf16 v16bf;
typedef __attribute__((ext_vector_type(8)))  float  v8f;

#define HID 256
#define TPB 256

// ---- native CDNA5 fp32 global atomic add (no-return, STOREcnt-tracked) ----
// Explicit device scope: RMW executes at the global L2 atomic units.
__device__ __forceinline__ void atom_add_f32(float* p, float v) {
  asm volatile("global_atomic_add_f32 %0, %1, off scope:SCOPE_DEV"
               :: "v"(p), "v"(v) : "memory");
}

union Q16 { uint4 q; bf16 b[8]; };

__global__ void k_zero_f4(float4* __restrict__ p, int n4) {
  int i = blockIdx.x * blockDim.x + threadIdx.x;
  if (i < n4) p[i] = make_float4(0.f, 0.f, 0.f, 0.f);
}

// fp32 -> bf16 cast, 8 elements per thread
__global__ void k_cast_bf16(const float* __restrict__ in, bf16* __restrict__ out, int n8) {
  int i = blockIdx.x * blockDim.x + threadIdx.x;
  if (i >= n8) return;
  const float4* ip = (const float4*)in + (size_t)i * 2;
  float4 a = ip[0], b = ip[1];
  Q16 u;
  u.b[0] = (bf16)a.x; u.b[1] = (bf16)a.y; u.b[2] = (bf16)a.z; u.b[3] = (bf16)a.w;
  u.b[4] = (bf16)b.x; u.b[5] = (bf16)b.y; u.b[6] = (bf16)b.z; u.b[7] = (bf16)b.w;
  ((uint4*)out)[i] = u.q;
}

// h0 = bf16(emb[x]); 32 threads per node, 8 features each
__global__ void k_gather_cast(const int* __restrict__ x, const float* __restrict__ emb,
                              bf16* __restrict__ out, int n) {
  int t = blockIdx.x * blockDim.x + threadIdx.x;
  int node = t >> 5, seg = t & 31;
  if (node >= n) return;
  const float4* ip = (const float4*)(emb + (size_t)x[node] * HID + seg * 8);
  float4 a = ip[0], b = ip[1];
  Q16 u;
  u.b[0] = (bf16)a.x; u.b[1] = (bf16)a.y; u.b[2] = (bf16)a.z; u.b[3] = (bf16)a.w;
  u.b[4] = (bf16)b.x; u.b[5] = (bf16)b.y; u.b[6] = (bf16)b.z; u.b[7] = (bf16)b.w;
  *(uint4*)(out + (size_t)node * HID + seg * 8) = u.q;
}

__global__ void k_deg(const int* __restrict__ dst, float* __restrict__ deg, int e) {
  int i = blockIdx.x * blockDim.x + threadIdx.x;
  if (i < e) atom_add_f32(deg + dst[i], 1.0f);
}

__global__ void k_inv_inplace(float* __restrict__ deg, int n) {
  int i = blockIdx.x * blockDim.x + threadIdx.x;
  if (i < n) deg[i] = 1.0f / fmaxf(deg[i], 1.0f);
}

// one wave per edge: 512B coalesced bf16 row read, 256 fp32 device-scope atomic adds
__global__ void k_scatter(const int* __restrict__ src, const int* __restrict__ dst,
                          const bf16* __restrict__ h, float* __restrict__ agg, int e) {
  int t = blockIdx.x * blockDim.x + threadIdx.x;   // E*32 = 102.4M < 2^31
  int edge = t >> 5, lane = t & 31;
  if (edge >= e) return;
  int s = src[edge], d = dst[edge];
  Q16 u; u.q = *(const uint4*)(h + (size_t)s * HID + lane * 8);
  float* ap = agg + (size_t)d * HID + lane * 8;
#pragma unroll
  for (int i = 0; i < 8; ++i) atom_add_f32(ap + i, (float)u.b[i]);
}

// mean = bf16(agg * inv[node])
__global__ void k_mean_cast(const float* __restrict__ agg, const float* __restrict__ inv,
                            bf16* __restrict__ out, int n) {
  int t = blockIdx.x * blockDim.x + threadIdx.x;
  int node = t >> 5, seg = t & 31;
  if (node >= n) return;
  float s = inv[node];
  const float4* ip = (const float4*)(agg + (size_t)node * HID + seg * 8);
  float4 a = ip[0], b = ip[1];
  Q16 u;
  u.b[0] = (bf16)(a.x * s); u.b[1] = (bf16)(a.y * s); u.b[2] = (bf16)(a.z * s); u.b[3] = (bf16)(a.w * s);
  u.b[4] = (bf16)(b.x * s); u.b[5] = (bf16)(b.y * s); u.b[6] = (bf16)(b.z * s); u.b[7] = (bf16)(b.w * s);
  *(uint4*)(out + (size_t)node * HID + seg * 8) = u.q;
}

// ---- WMMA operand loaders (ISA 7.12.2 layouts, wave32) ----
// A 16x32 bf16: lane&15 = row M; hi = lane>>4 selects K-halves.
// per-lane 16 values: K = kb+8*hi..+7  and  K = kb+16+8*hi..+7
__device__ __forceinline__ v16bf load_A16(const bf16* __restrict__ row, int kb, int hi) {
  union { uint4 q[2]; v16bf v; } u;
  u.q[0] = *(const uint4*)(row + kb + hi * 8);
  u.q[1] = *(const uint4*)(row + kb + 16 + hi * 8);
  return u.v;
}
// B 32x16 bf16 (row-striped across lanes): lane&15 = col N;
// per-lane 16 values: K = kb+8*hi..+7  and  K = kb+16+8*hi..+7 (same as A)
__device__ __forceinline__ v16bf load_B16(const bf16* __restrict__ row, int kb, int hi) {
  union { uint4 q[2]; v16bf v; } u;
  u.q[0] = *(const uint4*)(row + kb + hi * 8);
  u.q[1] = *(const uint4*)(row + kb + 16 + hi * 8);
  return u.v;
}

// out = act( A1 @ W1^T + [A2 @ W2^T] + bias ), A: [N][256] bf16, W: [256][256] bf16
// outb != null -> bf16 activation output; else fp32 to outf.
__global__ __launch_bounds__(TPB) void k_gemm(
    const bf16* __restrict__ A1, const bf16* __restrict__ W1,
    const bf16* __restrict__ A2, const bf16* __restrict__ W2,
    const float* __restrict__ bias, const float* __restrict__ slope,
    bf16* __restrict__ outb, float* __restrict__ outf, int nrows) {
  int wid  = (blockIdx.x * blockDim.x + threadIdx.x) >> 5;  // one 16-row M-tile per wave
  int lane = threadIdx.x & 31;
  int mtile = wid;
  if (mtile * 16 >= nrows) return;   // wave-uniform guard: EXEC stays all-ones for WMMA
  int r = lane & 15, hi = lane >> 4;

  v16bf a1[8], a2[8];
  const bf16* a1row = A1 + (size_t)(mtile * 16 + r) * HID;
#pragma unroll
  for (int ks = 0; ks < 8; ++ks) a1[ks] = load_A16(a1row, ks * 32, hi);
  const bool has2 = (A2 != nullptr);
  if (has2) {
    const bf16* a2row = A2 + (size_t)(mtile * 16 + r) * HID;
#pragma unroll
    for (int ks = 0; ks < 8; ++ks) a2[ks] = load_A16(a2row, ks * 32, hi);
  }
  const bool prelu = (slope != nullptr);
  float sl = prelu ? slope[0] : 0.0f;

  for (int nt = 0; nt < 16; ++nt) {
    v8f acc = {};
    const bf16* w1row = W1 + (size_t)(nt * 16 + r) * HID;
#pragma unroll
    for (int ks = 0; ks < 8; ++ks) {
      v16bf b = load_B16(w1row, ks * 32, hi);
      acc = __builtin_amdgcn_wmma_f32_16x16x32_bf16(false, a1[ks], false, b,
                                                    (short)0, acc, false, false);
    }
    if (has2) {
      const bf16* w2row = W2 + (size_t)(nt * 16 + r) * HID;
#pragma unroll
      for (int ks = 0; ks < 8; ++ks) {
        v16bf b = load_B16(w2row, ks * 32, hi);
        acc = __builtin_amdgcn_wmma_f32_16x16x32_bf16(false, a2[ks], false, b,
                                                      (short)0, acc, false, false);
      }
    }
    int col = nt * 16 + r;
    float bc = bias[col];
#pragma unroll
    for (int rr = 0; rr < 8; ++rr) {            // D VGPR rr -> row rr + 8*hi
      float v = acc[rr] + bc;
      if (prelu) v = (v >= 0.f) ? v : sl * v;
      size_t o = (size_t)(mtile * 16 + rr + hi * 8) * HID + col;
      if (outb) outb[o] = (bf16)v;
      else      outf[o] = v;
    }
  }
}

extern "C" void kernel_launch(void* const* d_in, const int* in_sizes, int n_in,
                              void* d_out, int out_size, void* d_ws, size_t ws_size,
                              hipStream_t stream) {
  (void)n_in; (void)out_size; (void)ws_size;
  const int N = in_sizes[0];
  const int E = in_sizes[1] / 2;
  const int*   x    = (const int*)d_in[0];
  const int*   src  = (const int*)d_in[1];
  const int*   dst  = src + E;
  const float* emb  = (const float*)d_in[3];
  const float* Wl[3] = {(const float*)d_in[4],  (const float*)d_in[8],  (const float*)d_in[12]};
  const float* bl[3] = {(const float*)d_in[5],  (const float*)d_in[9],  (const float*)d_in[13]};
  const float* Wr[3] = {(const float*)d_in[6],  (const float*)d_in[10], (const float*)d_in[14]};
  const float* al[3] = {(const float*)d_in[7],  (const float*)d_in[11], (const float*)d_in[15]};
  const float* Wout = (const float*)d_in[16];
  const float* bout = (const float*)d_in[17];

  // workspace layout (all 16B aligned; ~258 MB total)
  const size_t ND = (size_t)N * HID;
  float* agg = (float*)d_ws;           // ND f32
  float* deg = agg + ND;               // N f32 (becomes 1/max(deg,1))
  bf16*  hA  = (bf16*)(deg + N);       // ND bf16
  bf16*  hB  = hA + ND;                // ND bf16
  bf16*  mb  = hB + ND;                // ND bf16
  bf16*  wb  = mb + ND;                // 7 * 256*256 bf16

  auto blk = [](long n, int t) { return dim3((unsigned)((n + t - 1) / t)); };

  // weights -> bf16 once per call: order Wl1,Wr1,Wl2,Wr2,Wl3,Wr3,Wout
  const float* wsrc[7] = {Wl[0], Wr[0], Wl[1], Wr[1], Wl[2], Wr[2], Wout};
  for (int m = 0; m < 7; ++m)
    k_cast_bf16<<<blk(HID * HID / 8, TPB), TPB, 0, stream>>>(
        wsrc[m], wb + (size_t)m * HID * HID, HID * HID / 8);

  // h0 = bf16(emb[x])
  k_gather_cast<<<blk((long)N * 32, TPB), TPB, 0, stream>>>(x, emb, hA, N);

  // degree (graph-constant across layers), then invert in place
  k_zero_f4<<<blk(N / 4, TPB), TPB, 0, stream>>>((float4*)deg, N / 4);
  k_deg<<<blk(E, TPB), TPB, 0, stream>>>(dst, deg, E);
  k_inv_inplace<<<blk(N, TPB), TPB, 0, stream>>>(deg, N);

  const int mtiles = (N + 15) / 16;
  dim3 ggrid((unsigned)((mtiles + 7) / 8));  // 8 waves / block

  bf16* cur = hA; bf16* nxt = hB;
  for (int L = 0; L < 3; ++L) {
    k_zero_f4<<<blk((long)(ND / 4), TPB), TPB, 0, stream>>>((float4*)agg, (int)(ND / 4));
    k_scatter<<<blk((long)E * 32, TPB), TPB, 0, stream>>>(src, dst, cur, agg, E);
    k_mean_cast<<<blk((long)N * 32, TPB), TPB, 0, stream>>>(agg, deg, mb, N);
    k_gemm<<<ggrid, TPB, 0, stream>>>(mb, wb + (size_t)(2 * L) * HID * HID,
                                      cur, wb + (size_t)(2 * L + 1) * HID * HID,
                                      bl[L], al[L], nxt, nullptr, N);
    bf16* t = cur; cur = nxt; nxt = t;
  }
  // out = h3 @ Wout^T + bout (fp32)
  k_gemm<<<ggrid, TPB, 0, stream>>>(cur, wb + (size_t)6 * HID * HID, nullptr, nullptr,
                                    bout, nullptr, nullptr, (float*)d_out, N);
}
